// Payoff_Net_72438918414869
// MI455X (gfx1250) — matrix operand order repaired and checked by reading, not compiled, not run on hardware
//
#include <hip/hip_runtime.h>
#include <math.h>

typedef __attribute__((ext_vector_type(16))) _Float16 v16h;
typedef __attribute__((ext_vector_type(8)))  float    v8f;

#define ACTN    64
#define CTX     128
#define MTRI    2016          // 64*63/2
#define NTILES2 126           // 2016 / 16
#define TB      16            // batch rows per workgroup
#define BLOCK   256           // 8 wave32 waves
#define NITERS  100
#define PSTR    66            // padded floats per P row: bank = (2i+j)%64, conflict-free rows+cols
#define PBAT    (64*PSTR)     // 4224 floats per batch element
#define HSTR    136           // padded halfs per h row

#define SM_P 0
#define SM_U (TB*PBAT*4)                   // 270336
#define SM_V (SM_U + TB*64*4)              // 274432
#define SM_H (SM_V + TB*64*4)              // 278528
#define SMEM_BYTES (SM_H + TB*HSTR*2)      // 282880  (< 320 KB / WGP)

__global__ __launch_bounds__(BLOCK, 1)
void qre_fused_kernel(const float* __restrict__ x,  const float* __restrict__ W1,
                      const float* __restrict__ b1, const float* __restrict__ W2,
                      const float* __restrict__ b2, float* __restrict__ out)
{
    extern __shared__ char smem[];
    float*    P_ls = (float*)(smem + SM_P);
    float*    u_ls = (float*)(smem + SM_U);
    float*    v_ls = (float*)(smem + SM_V);
    _Float16* h_ls = (_Float16*)(smem + SM_H);

    const int tid  = threadIdx.x;
    const int w    = tid >> 5;        // wave id 0..7
    const int l    = tid & 31;        // lane in wave
    const int nl   = l & 15;          // N / M index within WMMA tile
    const int half = l >> 4;          // lane half (0: lanes 0-15, 1: lanes 16-31)
    const int b0   = blockIdx.x * TB;

    // zero P diagonal (off-diagonals are all written by the scatter below)
    for (int idx = tid; idx < TB*64; idx += BLOCK) {
        int bb = idx >> 6, d = idx & 63;
        P_ls[bb*PBAT + d*PSTR + d] = 0.0f;
    }
    // init u = v = 1/n
    for (int idx = tid; idx < TB*64; idx += BLOCK) {
        u_ls[idx] = 1.0f/64.0f;
        v_ls[idx] = 1.0f/64.0f;
    }

    // ---------------- GEMM1: h = LeakyReLU(x @ W1 + b1)  (wave w owns n-tile w)
    {
        v8f acc = {};
        #pragma unroll
        for (int kc = 0; kc < 4; ++kc) {
            v16h a, bfr;
            #pragma unroll
            for (int r = 0; r < 8; ++r) {
                // A 16x32 f16 layout: VGPR r pairs K = 2(r&3) + 16*(r>>2) + 8*half
                int ka = kc*32 + 2*(r&3) + ((r>>2)<<4) + half*8;
                const float* xp = x + (size_t)(b0 + nl)*CTX + ka;
                a[2*r]   = (_Float16)xp[0];
                a[2*r+1] = (_Float16)xp[1];
                // B 32x16 f16 layout: VGPR r pairs K = 2(r&3) + 16*(r>>2) + 8*half, col = nl
                int kb = kc*32 + 2*(r&3) + ((r>>2)<<4) + half*8;
                const float* wp = W1 + (size_t)kb*CTX + (w*16 + nl);
                bfr[2*r]   = (_Float16)wp[0];
                bfr[2*r+1] = (_Float16)wp[CTX];
            }
            acc = __builtin_amdgcn_wmma_f32_16x16x32_f16(false, a, false, bfr,
                                                         (short)0, acc, false, false);
        }
        int ncol = w*16 + nl;
        float bias = b1[ncol];
        #pragma unroll
        for (int r = 0; r < 8; ++r) {
            // C layout: VGPR r -> M = r + 8*half, N = nl
            float hv = acc[r] + bias;
            hv = (hv >= 0.0f) ? hv : 0.1f * hv;
            h_ls[(r + half*8)*HSTR + ncol] = (_Float16)hv;
        }
    }
    __syncthreads();

    // ---------------- GEMM2: p = h @ W2 + b2 ; scatter antisymmetric P into LDS
    {
        v16h afr[4];   // A fragments of h, reused across all n-tiles of this wave
        #pragma unroll
        for (int kc = 0; kc < 4; ++kc) {
            #pragma unroll
            for (int r = 0; r < 8; ++r) {
                int ka = kc*32 + 2*(r&3) + ((r>>2)<<4) + half*8;
                const _Float16* hp = h_ls + nl*HSTR + ka;
                afr[kc][2*r]   = hp[0];
                afr[kc][2*r+1] = hp[1];
            }
        }
        for (int nt = w; nt < NTILES2; nt += 8) {
            if (nt + 8 < NTILES2)
                __builtin_prefetch(W2 + (size_t)(nt + 8)*16 + nl, 0, 1);
            v8f acc = {};
            #pragma unroll
            for (int kc = 0; kc < 4; ++kc) {
                v16h bfr;
                #pragma unroll
                for (int r = 0; r < 8; ++r) {
                    int kb = kc*32 + 2*(r&3) + ((r>>2)<<4) + half*8;
                    const float* wp = W2 + (size_t)kb*MTRI + nt*16 + nl;
                    bfr[2*r]   = (_Float16)wp[0];
                    bfr[2*r+1] = (_Float16)wp[MTRI];
                }
                acc = __builtin_amdgcn_wmma_f32_16x16x32_f16(false, afr[kc], false, bfr,
                                                             (short)0, acc, false, false);
            }
            // strict-upper-tri column t -> (i, j), n = 64; offset(i) = i*(127-i)/2
            int t = nt*16 + nl;
            float bias = b2[t];
            int i = (int)floorf((127.0f - sqrtf(16129.0f - 8.0f*(float)t)) * 0.5f);
            if (i < 0)  i = 0;
            if (i > 62) i = 62;
            while (i > 0 && (i*(127 - i))/2 > t) --i;
            while (((i + 1)*(126 - i))/2 <= t) ++i;
            int j = i + 1 + (t - (i*(127 - i))/2);
            #pragma unroll
            for (int r = 0; r < 8; ++r) {
                float val = acc[r] + bias;                // p[b, t]
                float* Pb = P_ls + (size_t)(r + half*8)*PBAT;
                Pb[i*PSTR + j] =  val;                    // P[i][j] =  p
                Pb[j*PSTR + i] = -val;                    // P[j][i] = -p
            }
        }
    }
    __syncthreads();

    // ---------------- QRE fixed-point solver: wave w owns batch rows 2w, 2w+1
    for (int it = 0; it < NITERS; ++it) {
        // u = softmax(-P v)
        #pragma unroll
        for (int bb = 0; bb < 2; ++bb) {
            int b = 2*w + bb;
            const float* Pb = P_ls + (size_t)b*PBAT;
            const float* vv = v_ls + b*64;
            float a0 = 0.0f, a1 = 0.0f;          // rows l and l+32
            #pragma unroll 8
            for (int jj = 0; jj < 64; jj += 2) {
                float2 vj = *(const float2*)(vv + jj);                 // LDS broadcast
                float2 p0 = *(const float2*)(Pb + l*PSTR + jj);
                float2 p1 = *(const float2*)(Pb + (l + 32)*PSTR + jj);
                a0 = fmaf(p0.x, vj.x, fmaf(p0.y, vj.y, a0));
                a1 = fmaf(p1.x, vj.x, fmaf(p1.y, vj.y, a1));
            }
            a0 = -a0; a1 = -a1;
            float mx = fmaxf(a0, a1);
            #pragma unroll
            for (int off = 16; off > 0; off >>= 1)
                mx = fmaxf(mx, __shfl_xor(mx, off, 32));
            float e0 = __expf(a0 - mx), e1 = __expf(a1 - mx);
            float s = e0 + e1;
            #pragma unroll
            for (int off = 16; off > 0; off >>= 1)
                s += __shfl_xor(s, off, 32);
            float inv = 1.0f / s;
            u_ls[b*64 + l]      = e0 * inv;
            u_ls[b*64 + l + 32] = e1 * inv;
        }
        __syncthreads();
        // v = softmax(P^T u)
        #pragma unroll
        for (int bb = 0; bb < 2; ++bb) {
            int b = 2*w + bb;
            const float* Pb = P_ls + (size_t)b*PBAT;
            const float* uu = u_ls + b*64;
            float c0 = 0.0f, c1 = 0.0f;          // columns l and l+32
            #pragma unroll 8
            for (int ii = 0; ii < 64; ii += 2) {
                float2 ui = *(const float2*)(uu + ii);
                c0 = fmaf(Pb[ii*PSTR + l],          ui.x, c0);
                c1 = fmaf(Pb[ii*PSTR + l + 32],     ui.x, c1);
                c0 = fmaf(Pb[(ii+1)*PSTR + l],      ui.y, c0);
                c1 = fmaf(Pb[(ii+1)*PSTR + l + 32], ui.y, c1);
            }
            float mx = fmaxf(c0, c1);
            #pragma unroll
            for (int off = 16; off > 0; off >>= 1)
                mx = fmaxf(mx, __shfl_xor(mx, off, 32));
            float e0 = __expf(c0 - mx), e1 = __expf(c1 - mx);
            float s = e0 + e1;
            #pragma unroll
            for (int off = 16; off > 0; off >>= 1)
                s += __shfl_xor(s, off, 32);
            float inv = 1.0f / s;
            v_ls[b*64 + l]      = e0 * inv;
            v_ls[b*64 + l + 32] = e1 * inv;
        }
        __syncthreads();
    }

    // ---------------- write out: [b][0:64] = u, [b][64:128] = v
    for (int idx = tid; idx < TB*128; idx += BLOCK) {
        int bl = idx >> 7, col = idx & 127;
        float val = (col < 64) ? u_ls[bl*64 + col] : v_ls[bl*64 + (col - 64)];
        out[(size_t)(b0 + bl)*128 + col] = val;
    }
}

extern "C" void kernel_launch(void* const* d_in, const int* in_sizes, int n_in,
                              void* d_out, int out_size, void* d_ws, size_t ws_size,
                              hipStream_t stream) {
    const float* x  = (const float*)d_in[0];
    const float* W1 = (const float*)d_in[1];
    const float* b1 = (const float*)d_in[2];
    const float* W2 = (const float*)d_in[3];
    const float* b2 = (const float*)d_in[4];
    float* out = (float*)d_out;

    static_assert(SMEM_BYTES <= 320*1024, "LDS budget exceeds WGP capacity");
    (void)hipFuncSetAttribute((const void*)qre_fused_kernel,
                              hipFuncAttributeMaxDynamicSharedMemorySize,
                              SMEM_BYTES);

    int batch = in_sizes[0] / CTX;          // 8192
    dim3 grid(batch / TB), block(BLOCK);    // 512 workgroups x 256 threads
    qre_fused_kernel<<<grid, block, SMEM_BYTES, stream>>>(x, W1, b1, W2, b2, out);
}
